// GNN_LSTM_5574867551045
// MI455X (gfx1250) — matrix operand order repaired and checked by reading, not compiled, not run on hardware
//
#include <hip/hip_runtime.h>

typedef _Float16 v16h __attribute__((ext_vector_type(16)));
typedef _Float16 h8   __attribute__((ext_vector_type(8)));
typedef float    v8f  __attribute__((ext_vector_type(8)));

#define B_    32
#define T_    168
#define E_    384
#define NH    32      // hydro stations
#define NM    64      // meteo nodes
#define NN    96      // total nodes
#define DG    64      // HID_GNN
#define HL    128     // HID_LSTM
#define G4    512     // 4*HID_LSTM
#define KTOT  192     // DG + HL
#define FUT   24
#define SLOPE 0.01f

// ---------------- GNN stage ----------------

// agg[g, dst] += x[g, src]  (x = concat(hydro, meteo) along node dim, F=1)
__global__ void k_scatter(const float* __restrict__ meteo, const float* __restrict__ hydro,
                          const int* __restrict__ edges, float* __restrict__ agg) {
    const int g = blockIdx.x;                       // 0 .. B*T-1
    for (int e = threadIdx.x; e < E_; e += blockDim.x) {
        const int src = edges[e];
        const int dst = edges[E_ + e];
        const float v = (src < NH) ? hydro[(long)g * NH + src]
                                   : meteo[(long)g * NM + (src - NH)];
        atomicAdd(&agg[(long)g * NN + dst], v);
    }
}

// h = leaky(x*W_root + agg*W_rel + b) for hydro nodes; emit f16 in [s][t][b][64]
__global__ void k_gnn(const float* __restrict__ hydro, const float* __restrict__ agg,
                      const float* __restrict__ Wr, const float* __restrict__ Wl,
                      const float* __restrict__ bg, _Float16* __restrict__ X) {
    const long id = (long)blockIdx.x * blockDim.x + threadIdx.x;
    if (id >= (long)B_ * T_ * NH * DG) return;
    const int k = (int)(id & (DG - 1));
    const int n = (int)((id >> 6) & (NH - 1));
    const long g = id >> 11;                        // b*T + t
    const int b = (int)(g / T_), t = (int)(g % T_);
    const float x = hydro[g * NH + n];
    const float a = agg[g * NN + n];
    float v = x * Wr[k] + a * Wl[k] + bg[k];
    v = v > 0.f ? v : SLOPE * v;
    X[(((long)n * T_ + t) * B_ + b) * DG + k] = (_Float16)v;
}

// WT[s][n][k] (K-contiguous, f16): k<64 -> Wih[s][n][k], else Whh[s][n][k-64]
__global__ void k_wt(const float* __restrict__ Wih, const float* __restrict__ Whh,
                     _Float16* __restrict__ WT) {
    const long id = (long)blockIdx.x * blockDim.x + threadIdx.x;
    if (id >= (long)NH * G4 * KTOT) return;
    const int k = (int)(id % KTOT);
    const long nr = id / KTOT;
    const int n = (int)(nr % G4);
    const int s = (int)(nr / G4);
    const float v = (k < DG) ? Wih[((long)s * G4 + n) * DG + k]
                             : Whh[((long)s * G4 + n) * HL + (k - DG)];
    WT[id] = (_Float16)v;
}

// ---------------- LSTM stage (WMMA) ----------------
// 1 block = 1 station, 256 threads = 8 waves.
// Wave w owns gate-columns [q*128 + 16w, +16) for q=i,f,g,o -> cell math in regs.

__device__ __forceinline__ float sigm(float x) { return 1.f / (1.f + __expf(-x)); }

__global__ void __launch_bounds__(256) k_lstm(const _Float16* __restrict__ X,
                                              const _Float16* __restrict__ WT,
                                              const float* __restrict__ bih,
                                              const float* __restrict__ bhh,
                                              float* __restrict__ lastH) {
    __shared__ _Float16 hbuf[B_ * HL];              // 8 KB, h_t in f16
    const int s     = blockIdx.x;
    const int tid   = threadIdx.x;
    const int w     = tid >> 5;                     // wave 0..7
    const int lane  = tid & 31;
    const int lrow  = lane & 15;                    // A: row-in-tile; B/C: col-in-tile
    const int lhalf = lane >> 4;

    for (int i = tid; i < B_ * HL; i += 256) hbuf[i] = (_Float16)0.f;
    __syncthreads();

    // Per-lane gate biases (one column per gate for this lane)
    float bias[4];
#pragma unroll
    for (int q = 0; q < 4; ++q) {
        const int col = q * HL + w * 16 + lrow;
        bias[q] = bih[(long)s * G4 + col] + bhh[(long)s * G4 + col];
    }

    const _Float16* Xs = X  + (long)s * T_ * B_ * DG;
    const _Float16* Ws = WT + (long)s * G4 * KTOT;

    v8f acc[4][2];
    v8f cst[2]; cst[0] = {}; cst[1] = {};
    float hreg[2][8];
#pragma unroll
    for (int m = 0; m < 2; ++m)
#pragma unroll
        for (int v = 0; v < 8; ++v) hreg[m][v] = 0.f;

    for (int t = 0; t < T_; ++t) {
#pragma unroll
        for (int q = 0; q < 4; ++q) { acc[q][0] = {}; acc[q][1] = {}; }
        const _Float16* Xt = Xs + (long)t * B_ * DG;

#pragma unroll
        for (int ks = 0; ks < 6; ++ks) {            // K = 192 = 6 x 32
            const int k0 = ks * 32;
            // A fragments (16x32 f16): lane holds row lrow(+16m), K halves
            // [k0+8*lhalf .. +7] and [k0+8*lhalf+16 .. +7]
            v16h afrag[2];
#pragma unroll
            for (int m = 0; m < 2; ++m) {
                const int row = m * 16 + lrow;
                const int kb  = k0 + lhalf * 8;
                h8 lo, hi;
                if (k0 < DG) {                      // x_t portion from global f16
                    lo = *(const h8*)(Xt + (long)row * DG + kb);
                    hi = *(const h8*)(Xt + (long)row * DG + kb + 16);
                } else {                            // h portion from LDS
                    lo = *(const h8*)(&hbuf[row * HL + (kb - DG)]);
                    hi = *(const h8*)(&hbuf[row * HL + (kb - DG) + 16]);
                }
                afrag[m] = __builtin_shufflevector(lo, hi,
                            0,1,2,3,4,5,6,7,8,9,10,11,12,13,14,15);
            }
#pragma unroll
            for (int q = 0; q < 4; ++q) {
                // B fragment (32x16 f16): lane = col, K halves [k0+8*lhalf .. +7] and [k0+8*lhalf+16 .. +7]
                const int col = q * HL + w * 16 + lrow;
                const h8 blo = *(const h8*)(Ws + (long)col * KTOT + k0 + lhalf * 8);
                const h8 bhi = *(const h8*)(Ws + (long)col * KTOT + k0 + lhalf * 8 + 16);
                const v16h bfrag = __builtin_shufflevector(blo, bhi,
                            0,1,2,3,4,5,6,7,8,9,10,11,12,13,14,15);
                acc[q][0] = __builtin_amdgcn_wmma_f32_16x16x32_f16(
                    false, afrag[0], false, bfrag, (short)0, acc[q][0], false, false);
                acc[q][1] = __builtin_amdgcn_wmma_f32_16x16x32_f16(
                    false, afrag[1], false, bfrag, (short)0, acc[q][1], false, false);
            }
        }

        __syncthreads();                            // all reads of old h done
        const int ch = w * 16 + lrow;               // this lane's H column
#pragma unroll
        for (int m = 0; m < 2; ++m) {
#pragma unroll
            for (int v = 0; v < 8; ++v) {
                const int row = m * 16 + lhalf * 8 + v;   // batch index (C layout)
                const float ig = sigm(acc[0][m][v] + bias[0]);
                const float fg = sigm(acc[1][m][v] + bias[1]);
                const float gg = tanhf(acc[2][m][v] + bias[2]);
                const float og = sigm(acc[3][m][v] + bias[3]);
                const float c  = fg * cst[m][v] + ig * gg;
                cst[m][v] = c;
                const float h = og * tanhf(c);
                hreg[m][v] = h;
                hbuf[row * HL + ch] = (_Float16)h;
            }
        }
        __syncthreads();                            // new h visible next step
    }

    // last hidden, f32, [s][b][128]
#pragma unroll
    for (int m = 0; m < 2; ++m)
#pragma unroll
        for (int v = 0; v < 8; ++v) {
            const int row = m * 16 + lhalf * 8 + v;
            lastH[((long)s * B_ + row) * HL + (w * 16 + lrow)] = hreg[m][v];
        }
}

// ---------------- head ----------------
__global__ void k_head(const float* __restrict__ lastH, const float* __restrict__ Wlin,
                       const float* __restrict__ blin, float* __restrict__ out) {
    const int id = blockIdx.x * blockDim.x + threadIdx.x;
    if (id >= B_ * NH * FUT) return;
    const int f = id % FUT;
    const int s = (id / FUT) % NH;
    const int b = id / (FUT * NH);
    const float* hp = lastH + ((long)s * B_ + b) * HL;
    const float* wp = Wlin + (long)f * HL;
    float acc = blin[f];
#pragma unroll 8
    for (int k = 0; k < HL; ++k) acc += hp[k] * wp[k];
    acc = acc > 0.f ? acc : SLOPE * acc;
    out[((long)b * NH + s) * FUT + f] = acc;
}

// ---------------- launch ----------------
extern "C" void kernel_launch(void* const* d_in, const int* in_sizes, int n_in,
                              void* d_out, int out_size, void* d_ws, size_t ws_size,
                              hipStream_t stream) {
    const float* meteo = (const float*)d_in[0];
    const float* hydro = (const float*)d_in[1];
    const int*   edges = (const int*)d_in[2];
    const float* Wr    = (const float*)d_in[3];
    const float* Wl    = (const float*)d_in[4];
    const float* bg    = (const float*)d_in[5];
    const float* Wih   = (const float*)d_in[6];
    const float* Whh   = (const float*)d_in[7];
    const float* bihp  = (const float*)d_in[8];
    const float* bhhp  = (const float*)d_in[9];
    const float* Wlin  = (const float*)d_in[10];
    const float* blin  = (const float*)d_in[11];
    float* out = (float*)d_out;

    const long G = (long)B_ * T_;                       // 5376
    char* ws = (char*)d_ws;
    const size_t off_agg  = 0;                          // 5376*96*4   = 2,064,384 B
    const size_t off_X    = off_agg + 2064384;          // 32*168*32*64*2 = 22,020,096 B
    const size_t off_WT   = off_X   + 22020096;         // 32*512*192*2   = 6,291,456 B
    const size_t off_last = off_WT  + 6291456;          // 32*32*128*4    =   524,288 B
    float*    agg   = (float*)(ws + off_agg);
    _Float16* X     = (_Float16*)(ws + off_X);
    _Float16* WT    = (_Float16*)(ws + off_WT);
    float*    lastH = (float*)(ws + off_last);

    hipMemsetAsync(agg, 0, (size_t)G * NN * sizeof(float), stream);
    k_scatter<<<(int)G, 128, 0, stream>>>(meteo, hydro, edges, agg);

    { long n = G * NH * DG;
      k_gnn<<<(int)((n + 255) / 256), 256, 0, stream>>>(hydro, agg, Wr, Wl, bg, X); }
    { long n = (long)NH * G4 * KTOT;
      k_wt<<<(int)((n + 255) / 256), 256, 0, stream>>>(Wih, Whh, WT); }

    k_lstm<<<NH, 256, 0, stream>>>(X, WT, bihp, bhhp, lastH);

    { int n = B_ * NH * FUT;
      k_head<<<(n + 255) / 256, 256, 0, stream>>>(lastH, Wlin, blin, out); }
}
